// AwkwardNN_55568286875783
// MI455X (gfx1250) — matrix operand order repaired and checked by reading, not compiled, not run on hardware
//
#include <hip/hip_runtime.h>
#include <math.h>

#define H      2048
#define FAN    2049
#define DEPTH  16
#define TTOT   4096
#define OUTSZ  128

#define NB     64          // persistent blocks (each owns 32 output columns)
#define COLS   32
#define NWAVE  8           // 256 threads, wave32

// LDS layout (dynamic):
//  [0 .. 131071]        B operands, 2 tiles * 64KB.
//     per tile: chunk0 region 32KB then chunk1 region 32KB;
//     within a region: (kb*32 + lane)*16 bytes  (kb = K-block 0..63)
//  [131072 .. 135167]   h as bf16 (2048 * 2B) for A operands (async-loaded)
//  [135168 .. 136191]   cross-wave reduction: 2 tiles * 8 waves * 16 floats
#define B_TILE   65536u
#define B_CHUNK  32768u
#define HBF_OFF  131072u
#define RED_OFF  135168u
#define LDS_BYTES 136192u

typedef __attribute__((ext_vector_type(16))) __bf16 v16bf;
typedef __attribute__((ext_vector_type(8)))  float  v8f;
typedef __attribute__((ext_vector_type(4)))  float  f32x4;

union BV { f32x4 q[2]; v16bf v; };

__device__ __forceinline__ unsigned short f2bf(float f) {
    unsigned u = __builtin_bit_cast(unsigned, f);
    u += 0x7FFFu + ((u >> 16) & 1u);           // round-to-nearest-even
    return (unsigned short)(u >> 16);
}

__device__ __forceinline__ void wait_asynccnt0() {
#if __has_builtin(__builtin_amdgcn_s_wait_asynccnt)
    __builtin_amdgcn_s_wait_asynccnt(0);
#else
    asm volatile("s_wait_asynccnt 0x0" ::: "memory");
#endif
}

__device__ __forceinline__ void grid_barrier(unsigned* cnt, unsigned* gen) {
    __threadfence();                            // make this block's stores visible device-wide
    __syncthreads();
    if (threadIdx.x == 0) {
        unsigned g = __hip_atomic_load(gen, __ATOMIC_RELAXED, __HIP_MEMORY_SCOPE_AGENT);
        unsigned a = __hip_atomic_fetch_add(cnt, 1u, __ATOMIC_ACQ_REL, __HIP_MEMORY_SCOPE_AGENT);
        if (a == NB - 1) {
            __hip_atomic_store(cnt, 0u, __ATOMIC_RELAXED, __HIP_MEMORY_SCOPE_AGENT);
            __hip_atomic_store(gen, g + 1u, __ATOMIC_RELEASE, __HIP_MEMORY_SCOPE_AGENT);
        } else {
            while (__hip_atomic_load(gen, __ATOMIC_ACQUIRE, __HIP_MEMORY_SCOPE_AGENT) == g) {
                __builtin_amdgcn_s_sleep(2);
            }
        }
    }
    __syncthreads();
    __threadfence();
}

extern "C" __global__ void awkward_rnn_kernel(
    const float* __restrict__ x, const float* __restrict__ h0,
    const float* __restrict__ W, const float* __restrict__ bvec,
    const float* __restrict__ Wout, const float* __restrict__ bout,
    const int* __restrict__ mk,
    float* __restrict__ out,
    float* __restrict__ hb0, float* __restrict__ hb1,
    unsigned short* __restrict__ hbf0, unsigned short* __restrict__ hbf1,
    unsigned* __restrict__ ctrl)
{
    extern __shared__ __align__(16) unsigned char lds[];
    const int tid  = threadIdx.x;
    const int bid  = blockIdx.x;
    const int n0   = bid * COLS;
    const int wave = tid >> 5;
    const int lane = tid & 31;

    // LDS byte offset (within this block's allocation) of this thread's async slot
    const unsigned hbf_slot = (unsigned)(uintptr_t)(lds + HBF_OFF + tid * 16);

    // --- markers -> cumulative counts, total step count (uniform across blocks) ---
    int cum[DEPTH];
    { int c = 0;
      for (int i = 0; i < DEPTH; ++i) { c += mk[i]; cum[i] = c; } }
    int steps = cum[DEPTH - 1]; if (steps > TTOT) steps = TTOT;

    // --- init hidden buffers (fp32 master + bf16 mirror), each block its slice ---
    if (tid < COLS) {
        float hv = h0[n0 + tid];
        hb0[n0 + tid]  = hv;
        hbf0[n0 + tid] = f2bf(hv);
    }
    grid_barrier(ctrl, ctrl + 1);

    int lid = 0;
    int cached = -1;

    for (int t = 0; t < steps; ++t) {
        while (lid < DEPTH - 1 && cum[lid] <= t) ++lid;

        const unsigned short* curbf = (t & 1) ? hbf1 : hbf0;
        float*                nxt   = (t & 1) ? hb0  : hb1;
        unsigned short*       nxtbf = (t & 1) ? hbf0 : hbf1;
        const float*          Wd    = W + (size_t)lid * FAN * H;

        // --- kick off async broadcast of bf16 hidden into LDS (one B128 per thread) ---
        {
            unsigned long long gaddr = (unsigned long long)(uintptr_t)curbf
                                     + (unsigned long long)tid * 16ull;
            asm volatile("global_load_async_to_lds_b128 %0, %1, off"
                         :: "v"(hbf_slot), "v"(gaddr)
                         : "memory");
        }

        // --- (re)load + bf16-swizzle this block's weight slice on layer change ---
        if (lid != cached) {
            cached = lid;
            for (int it = 0; it < 256; ++it) {
                int idx = tid + (it << 8);          // 0 .. 65535
                int k   = idx >> 5;                 // 0 .. 2047 (weight row 1+k)
                int c   = idx & 31;                 // column within slice
                float wv = Wd[(size_t)(1 + k) * H + (n0 + c)];
                int kb = k >> 5, kl = k & 31;
                int l  = (c & 15) + (kl & 16);      // B lane
                int j  = kl & 15;                   // element within lane group
                unsigned off = (unsigned)(c >> 4) * B_TILE + (unsigned)(j >> 3) * B_CHUNK
                             + ((unsigned)(kb * 32 + l) << 4) + ((unsigned)(j & 7) << 1);
                *(unsigned short*)(lds + off) = f2bf(wv);
            }
        }

        wait_asynccnt0();                           // this wave's 512B slice landed in LDS
        __syncthreads();                            // -> whole bf16 hidden visible

        // --- WMMA matvec: each wave covers K = [wave*256, wave*256+256) for both tiles ---
        v8f acc0 = {};
        v8f acc1 = {};
        for (int i = 0; i < 8; ++i) {
            int kb = wave * 8 + i;
            BV av, b0, b1;
            unsigned ab = HBF_OFF + (unsigned)(kb << 6) + (unsigned)((lane >> 4) << 5);
            av.q[0] = *(const f32x4*)(lds + ab);
            av.q[1] = *(const f32x4*)(lds + ab + 16);
            unsigned bb = (unsigned)((kb * 32 + lane) << 4);
            b0.q[0] = *(const f32x4*)(lds + bb);
            b0.q[1] = *(const f32x4*)(lds + bb + B_CHUNK);
            b1.q[0] = *(const f32x4*)(lds + B_TILE + bb);
            b1.q[1] = *(const f32x4*)(lds + B_TILE + bb + B_CHUNK);
            acc0 = __builtin_amdgcn_wmma_f32_16x16x32_bf16(false, av.v, false, b0.v,
                                                           (short)0, acc0, false, false);
            acc1 = __builtin_amdgcn_wmma_f32_16x16x32_bf16(false, av.v, false, b1.v,
                                                           (short)0, acc1, false, false);
        }
        // row M=0 result: lanes 0..15, acc[0] = column (lane)
        float* red = (float*)(lds + RED_OFF);
        if (lane < 16) {
            red[(0 * NWAVE + wave) * 16 + lane] = acc0[0];
            red[(1 * NWAVE + wave) * 16 + lane] = acc1[0];
        }
        __syncthreads();

        // --- combine: bias + x_t * W_row0 + relu; write fp32 master + bf16 mirror ---
        if (tid < COLS) {
            int tile = tid >> 4, n = tid & 15;
            float s = 0.f;
            for (int w = 0; w < NWAVE; ++w) s += red[(tile * NWAVE + w) * 16 + n];
            s += bvec[(size_t)lid * H + n0 + tid];
            s += x[t] * Wd[n0 + tid];               // fan-in row 0 (the scalar input)
            s = fmaxf(s, 0.f);
            nxt[n0 + tid]   = s;
            nxtbf[n0 + tid] = f2bf(s);
        }
        grid_barrier(ctrl, ctrl + 1);
    }

    const float* hfin = (steps & 1) ? hb1 : hb0;

    // --- write final hidden into out[128 .. 2175] (tuple order: output, hidden) ---
    if (tid < COLS) out[OUTSZ + n0 + tid] = hfin[n0 + tid];

    // --- block 0: logits + log_softmax into out[0 .. 127] ---
    if (bid == 0) {
        float* sl = (float*)lds;                    // reuse LDS as scratch
        float logit = 0.f;
        if (tid < OUTSZ) {
            logit = bout[tid];
            for (int k = 0; k < H; ++k) logit += hfin[k] * Wout[(size_t)k * OUTSZ + tid];
            sl[tid] = logit;
        }
        __syncthreads();
        if (tid == 0) {
            float mx = sl[0];
            for (int i = 1; i < OUTSZ; ++i) mx = fmaxf(mx, sl[i]);
            float se = 0.f;
            for (int i = 0; i < OUTSZ; ++i) se += expf(sl[i] - mx);
            sl[OUTSZ] = mx; sl[OUTSZ + 1] = logf(se);
        }
        __syncthreads();
        if (tid < OUTSZ) out[tid] = logit - sl[OUTSZ] - sl[OUTSZ + 1];
    }
}

extern "C" void kernel_launch(void* const* d_in, const int* in_sizes, int n_in,
                              void* d_out, int out_size, void* d_ws, size_t ws_size,
                              hipStream_t stream) {
    (void)in_sizes; (void)n_in; (void)out_size; (void)ws_size;
    const float* x    = (const float*)d_in[0];
    const float* h0   = (const float*)d_in[1];
    const float* W    = (const float*)d_in[2];
    const float* b    = (const float*)d_in[3];
    const float* Wout = (const float*)d_in[4];
    const float* bout = (const float*)d_in[5];
    const int*   mk   = (const int*)d_in[6];
    float* out = (float*)d_out;

    unsigned char* ws = (unsigned char*)d_ws;
    unsigned* ctrl = (unsigned*)ws;                 // [0]=counter, [1]=generation
    float*          hb0  = (float*)(ws + 256);                       // 8KB
    float*          hb1  = (float*)(ws + 256 + 8192);                // 8KB
    unsigned short* hbf0 = (unsigned short*)(ws + 256 + 16384);      // 4KB
    unsigned short* hbf1 = (unsigned short*)(ws + 256 + 20480);      // 4KB

    hipMemsetAsync(ctrl, 0, 256, stream);           // graph-capture-safe per-call reset
    hipLaunchKernelGGL(awkward_rnn_kernel, dim3(NB), dim3(256), LDS_BYTES, stream,
                       x, h0, W, b, Wout, bout, mk, out,
                       hb0, hb1, hbf0, hbf1, ctrl);
}
